// MEB_56195352101419
// MI455X (gfx1250) — compile-verified, matrix operand order checked by reading, not run on hardware
//
#include <hip/hip_runtime.h>

// ---------------------------------------------------------------------------
// MEB forward for MI455X (gfx1250, wave32). All 1x1 convs -> WMMA f16 GEMMs
// with f32 accumulation; activations kept f32 NHWC; depthwise focal convs are
// VALU/vmem. Weights pre-swizzled into the CDNA5 WMMA B-operand layout and
// staged into LDS per-workgroup via the Tensor Data Mover (tensor_load_to_lds).
// ---------------------------------------------------------------------------

#define B_   8
#define C_   64
#define H_   192
#define W_   192
#define HW_  (H_ * W_)       // 36864
#define P_   (B_ * HW_)      // 294912 pixels
#define H2_  96
#define W2_  96
#define HW2_ (H2_ * W2_)     // 9216
#define P2_  (B_ * HW2_)     // 73728

typedef __attribute__((ext_vector_type(16))) _Float16 v16h;
typedef __attribute__((ext_vector_type(8)))  float    v8f;
typedef __attribute__((ext_vector_type(4)))  unsigned int u32x4;
typedef __attribute__((ext_vector_type(8)))  unsigned int u32x8;

__device__ __forceinline__ float geluf(float x) {
  return 0.5f * x * (1.0f + erff(x * 0.7071067811865475f));
}

// ---------------------------------------------------------------------------
// LayerNorm over channels, NCHW input -> NHWC output (thread = pixel).
// ---------------------------------------------------------------------------
__global__ void k_ln_nchw2nhwc(const float* __restrict__ x,
                               const float* __restrict__ w,
                               const float* __restrict__ bp,
                               float* __restrict__ out) {
  int p = blockIdx.x * blockDim.x + threadIdx.x;
  if (p >= P_) return;
  int b = p / HW_, hw = p % HW_;
  const float* px = x + (size_t)b * C_ * HW_ + hw;
  float v[C_];
  float mu = 0.0f;
#pragma unroll
  for (int c = 0; c < C_; ++c) { v[c] = px[(size_t)c * HW_]; mu += v[c]; }
  mu *= (1.0f / C_);
  float var = 0.0f;
#pragma unroll
  for (int c = 0; c < C_; ++c) { float d = v[c] - mu; var += d * d; }
  var *= (1.0f / C_);
  float inv = rsqrtf(var + 1e-6f);
  float* o = out + (size_t)p * C_;
#pragma unroll
  for (int c = 0; c < C_; ++c) o[c] = w[c] * (v[c] - mu) * inv + bp[c];
}

// LayerNorm over channels, NHWC input (stride is_) -> NHWC output (stride 64).
__global__ void k_ln_nhwc(const float* __restrict__ x, int is_,
                          const float* __restrict__ w,
                          const float* __restrict__ bp,
                          float* __restrict__ out) {
  int p = blockIdx.x * blockDim.x + threadIdx.x;
  if (p >= P_) return;
  const float* px = x + (size_t)p * is_;
  float v[C_];
  float mu = 0.0f;
#pragma unroll
  for (int c = 0; c < C_; ++c) { v[c] = px[c]; mu += v[c]; }
  mu *= (1.0f / C_);
  float var = 0.0f;
#pragma unroll
  for (int c = 0; c < C_; ++c) { float d = v[c] - mu; var += d * d; }
  var *= (1.0f / C_);
  float inv = rsqrtf(var + 1e-6f);
  float* o = out + (size_t)p * C_;
#pragma unroll
  for (int c = 0; c < C_; ++c) o[c] = w[c] * (v[c] - mu) * inv + bp[c];
}

// ---------------------------------------------------------------------------
// Pre-swizzle f32 [Cout][Cin] weight into WMMA B-operand f16 tiles.
// Tile (nt, ks) is 512 f16: lane holds column n = nt*16+(lane&15),
// K set = ks*32 + khalf*8 + {0..7, 16..23}, matching the A-side loader below.
// ---------------------------------------------------------------------------
__global__ void k_prep_w(const float* __restrict__ src, _Float16* __restrict__ dst,
                         int Cout, int Cin, int ntiles, int ksteps) {
  int tile = blockIdx.x;                  // nt*ksteps + ks
  int nt = tile / ksteps, ks = tile % ksteps;
  int lane = threadIdx.x;                 // 0..31
  int n = nt * 16 + (lane & 15);
  int khalf = lane >> 4;
  _Float16* d = dst + (size_t)tile * 512 + lane * 16;
  for (int j = 0; j < 16; ++j) {
    int k = ks * 32 + khalf * 8 + ((j >> 3) << 4) + (j & 7);
    float v = (n < Cout && k < Cin) ? src[(size_t)n * Cin + k] : 0.0f;
    d[j] = (_Float16)v;
  }
}

// ---------------------------------------------------------------------------
// Generic 1x1-conv pixel-GEMM on WMMA. One wave per 16-pixel M-tile.
// Weight tiles staged global->LDS once per workgroup via the TDM
// (tensor_load_to_lds, TENSORcnt), then broadcast-read from LDS.
// KSTEPS/NTILES are compile-time so every WMMA register index is static.
// Fused epilogue: bias, GELU, per-batch cof scaling, residuals, NCHW store.
// ---------------------------------------------------------------------------
#define CF_GELU     1
#define CF_ACCUM    2
#define CF_SCALE    4
#define CF_ADD_NCHW 8
#define CF_OUT_NCHW 16
#define CF_ADD_NHWC 32

template <int KSTEPS, int NTILES>
__global__ void __launch_bounds__(256)
k_conv1x1_wmma(const float* __restrict__ in, int is_, int io_, int Cin,
               const _Float16* __restrict__ wt,
               const float* __restrict__ bias, int Cout,
               float* __restrict__ out, int os_, int oo_,
               int flags, const float* __restrict__ cof, int cof_idx,
               const float* __restrict__ resid, int rs_, int ro_) {
  __shared__ _Float16 sW[NTILES * KSTEPS * 512];

  // ---- Stage all weight tiles into LDS with the Tensor Data Mover ----
  if (threadIdx.x < 32) {                       // wave 0 only (wave-uniform)
    const unsigned ndw = (unsigned)(NTILES * KSTEPS * 512 / 2);  // dwords
    unsigned long long ga = (unsigned long long)(uintptr_t)wt;
    unsigned lds = (unsigned)(uintptr_t)(void*)&sW[0];           // LDS offset
    // D# group0: count=1 | lds_addr | global_addr[56:0] | type=2
    u32x4 g0;
    g0[0] = 1u;
    g0[1] = lds;
    g0[2] = (unsigned)ga;
    g0[3] = (unsigned)((ga >> 32) & 0x01FFFFFFu) | (2u << 30);
    // D# group1: data_size=4B; 1 row of ndw dwords; tile == tensor
    u32x8 g1;
    g1[0] = (2u << 16);                        // data_size = 2 (4 bytes)
    g1[1] = (ndw & 0xFFFFu) << 16;             // tensor_dim0[15:0]
    g1[2] = ((ndw >> 16) & 0xFFFFu) | (1u << 16);  // tensor_dim0[31:16] | tensor_dim1=1
    g1[3] = (ndw & 0xFFFFu) << 16;             // tile_dim0 = ndw (fits 16b)
    g1[4] = 1u;                                 // tile_dim1 = 1
    g1[5] = ndw;                                // tensor_dim0_stride
    g1[6] = 0u;
    g1[7] = 0u;
    asm volatile("tensor_load_to_lds %0, %1" :: "s"(g0), "s"(g1) : "memory");
    __builtin_amdgcn_s_wait_tensorcnt(0);
  }
  __syncthreads();

  int wave = (blockIdx.x * blockDim.x + threadIdx.x) >> 5;   // exact grid: P_/16 waves
  int lane = threadIdx.x & 31;
  int m0 = wave << 4;
  int khalf = lane >> 4;
  size_t prow = (size_t)(m0 + (lane & 15));

  // A operand: rows = pixels, K = channels (f32 -> f16 on the fly)
  v16h a[KSTEPS];
  const float* src = in + prow * (size_t)is_ + io_;
#pragma unroll
  for (int ks = 0; ks < KSTEPS; ++ks) {
#pragma unroll
    for (int j = 0; j < 16; ++j) {
      int c = ks * 32 + khalf * 8 + ((j >> 3) << 4) + (j & 7);
      float v = (c < Cin) ? src[c] : 0.0f;
      a[ks][j] = (_Float16)v;
    }
  }

#pragma unroll
  for (int nt = 0; nt < NTILES; ++nt) {
    v8f acc = {};
#pragma unroll
    for (int ks = 0; ks < KSTEPS; ++ks) {
      v16h bm = *(const v16h*)(&sW[(nt * KSTEPS + ks) * 512 + lane * 16]);
      acc = __builtin_amdgcn_wmma_f32_16x16x32_f16(
          false, a[ks], false, bm, (short)0, acc, false, false);
    }
    int n = nt * 16 + (lane & 15);
    bool nok = (n < Cout);
#pragma unroll
    for (int r = 0; r < 8; ++r) {
      if (!nok) continue;
      size_t pp = (size_t)(m0 + (khalf << 3) + r);
      int b = (int)(pp / HW_);
      int hw = (int)(pp % HW_);
      float v = acc[r] + bias[n];
      if (flags & CF_GELU)     v = geluf(v);
      if (flags & CF_SCALE)    v *= cof[b * 6 + cof_idx];
      if (flags & CF_ADD_NCHW) v += resid[((size_t)b * Cout + n) * HW_ + hw];
      if (flags & CF_ADD_NHWC) v += resid[pp * (size_t)rs_ + ro_ + n];
      if (flags & CF_OUT_NCHW) {
        out[((size_t)b * Cout + n) * HW_ + hw] = v;
      } else {
        size_t oi = pp * (size_t)os_ + oo_ + n;
        if (flags & CF_ACCUM) out[oi] += v; else out[oi] = v;
      }
    }
  }
}

// ---------------------------------------------------------------------------
// Depthwise KxK conv, NHWC, optional bias + GELU. thread = (pixel, channel).
// ---------------------------------------------------------------------------
__global__ void k_dwconv(const float* __restrict__ in, int is_, int io_,
                         const float* __restrict__ w, const float* __restrict__ bias,
                         float* __restrict__ out, int os_, int oo_,
                         int Cd, int Hd, int Wd, int K, int do_gelu, size_t total) {
  size_t idx = (size_t)blockIdx.x * blockDim.x + threadIdx.x;
  if (idx >= total) return;
  int c = (int)(idx % Cd);
  size_t p = idx / Cd;
  int xw = (int)(p % Wd);
  size_t t = p / Wd;
  int yh = (int)(t % Hd);
  int b = (int)(t / Hd);
  int pad = K >> 1;
  const float* wk = w + (size_t)c * K * K;
  float acc = 0.0f;
  for (int ky = 0; ky < K; ++ky) {
    int yy = yh + ky - pad;
    if (yy < 0 || yy >= Hd) continue;
    for (int kx = 0; kx < K; ++kx) {
      int xx = xw + kx - pad;
      if (xx < 0 || xx >= Wd) continue;
      acc += in[(((size_t)b * Hd + yy) * Wd + xx) * (size_t)is_ + io_ + c] * wk[ky * K + kx];
    }
  }
  if (bias) acc += bias[c];
  if (do_gelu) acc = geluf(acc);
  out[p * (size_t)os_ + oo_ + c] = acc;
}

// out (+)= in * gates[:, gl] (gate broadcast over channels)
__global__ void k_mulgate(const float* __restrict__ in, int is_, int io_,
                          const float* __restrict__ gate, int gs_, int gl,
                          float* __restrict__ out, int os_, int oo_,
                          int Cd, int accum, size_t total) {
  size_t idx = (size_t)blockIdx.x * blockDim.x + threadIdx.x;
  if (idx >= total) return;
  int c = (int)(idx % Cd);
  size_t p = idx / Cd;
  float v = in[p * (size_t)is_ + io_ + c] * gate[p * (size_t)gs_ + gl];
  size_t oi = p * (size_t)os_ + oo_ + c;
  if (accum) out[oi] += v; else out[oi] = v;
}

// elementwise product of two NHWC tensors
__global__ void k_ewmul(const float* __restrict__ a, int as_, int ao_,
                        const float* __restrict__ bsrc, int bs_, int bo_,
                        float* __restrict__ out, int os_, int oo_,
                        int Cd, size_t total) {
  size_t idx = (size_t)blockIdx.x * blockDim.x + threadIdx.x;
  if (idx >= total) return;
  int c = (int)(idx % Cd);
  size_t p = idx / Cd;
  out[p * (size_t)os_ + oo_ + c] =
      a[p * (size_t)as_ + ao_ + c] * bsrc[p * (size_t)bs_ + bo_ + c];
}

// spatial mean per (b, c): block per (b,c)
__global__ void k_rowmean(const float* __restrict__ in, int is_, int io_,
                          int Cd, int npix, float* __restrict__ out) {
  int bc = blockIdx.x;
  int b = bc / Cd, c = bc % Cd;
  const float* src = in + ((size_t)b * npix) * (size_t)is_ + io_ + c;
  float s = 0.0f;
  for (int i = threadIdx.x; i < npix; i += blockDim.x) s += src[(size_t)i * is_];
  __shared__ float red[256];
  red[threadIdx.x] = s;
  __syncthreads();
  for (int st = 128; st > 0; st >>= 1) {
    if ((int)threadIdx.x < st) red[threadIdx.x] += red[threadIdx.x + st];
    __syncthreads();
  }
  if (threadIdx.x == 0) out[bc] = red[0] / (float)npix;
}

// out = (acc + gelu(mean[b,c]) * gate[:,gl]) [* q] [* cof[b,ci]]
#define MB_MULQ 1
#define MB_COF  2
__global__ void k_combine(const float* __restrict__ acc, int as_, int ao_,
                          const float* __restrict__ mean,
                          const float* __restrict__ gate, int gs_, int gl,
                          const float* __restrict__ q, int qs_, int qo_,
                          const float* __restrict__ cof, int ci,
                          float* __restrict__ out, int os_, int oo_,
                          int Cd, int npix, int flags, size_t total) {
  size_t idx = (size_t)blockIdx.x * blockDim.x + threadIdx.x;
  if (idx >= total) return;
  int c = (int)(idx % Cd);
  size_t p = idx / Cd;
  int b = (int)(p / npix);
  float v = acc[p * (size_t)as_ + ao_ + c] +
            geluf(mean[b * Cd + c]) * gate[p * (size_t)gs_ + gl];
  if (flags & MB_MULQ) v *= q[p * (size_t)qs_ + qo_ + c];
  if (flags & MB_COF)  v *= cof[b * 6 + ci];
  out[p * (size_t)os_ + oo_ + c] = v;
}

// 2x2 average pool, NHWC (Hd, Wd = OUTPUT dims)
__global__ void k_avgpool2(const float* __restrict__ in, int is_, int io_,
                           float* __restrict__ out, int os_, int oo_,
                           int Cd, int Hd, int Wd, size_t total) {
  size_t idx = (size_t)blockIdx.x * blockDim.x + threadIdx.x;
  if (idx >= total) return;
  int c = (int)(idx % Cd);
  size_t p = idx / Cd;
  int wo = (int)(p % Wd);
  size_t t = p / Wd;
  int ho = (int)(t % Hd);
  int b = (int)(t / Hd);
  int Hi = Hd * 2, Wi = Wd * 2;
  size_t i00 = (((size_t)b * Hi + 2 * ho) * Wi + 2 * wo) * (size_t)is_ + io_ + c;
  size_t i10 = i00 + (size_t)Wi * is_;
  float v = 0.25f * (in[i00] + in[i00 + is_] + in[i10] + in[i10 + is_]);
  out[p * (size_t)os_ + oo_ + c] = v;
}

// out += q * bilerp_align_corners(hsrc) * cof[b,ci]   (half-res -> full-res)
__global__ void k_upsample_acc(const float* __restrict__ hsrc, int hs_,
                               const float* __restrict__ q, int qs_, int qo_,
                               const float* __restrict__ cof, int ci,
                               float* __restrict__ out, int os_, size_t total) {
  size_t idx = (size_t)blockIdx.x * blockDim.x + threadIdx.x;
  if (idx >= total) return;
  int c = (int)(idx % C_);
  size_t p = idx / C_;
  int xw = (int)(p % W_);
  size_t t = p / W_;
  int yh = (int)(t % H_);
  int b = (int)(t / H_);
  float ys = yh * ((float)(H2_ - 1) / (float)(H_ - 1));
  float xs = xw * ((float)(W2_ - 1) / (float)(W_ - 1));
  int y0 = (int)floorf(ys); int y1 = min(y0 + 1, H2_ - 1); float wy = ys - y0;
  int x0 = (int)floorf(xs); int x1 = min(x0 + 1, W2_ - 1); float wx = xs - x0;
  size_t base = (size_t)b * HW2_;
  const float* s = hsrc;
#define HS(yy, xx) s[(base + (size_t)(yy) * W2_ + (xx)) * (size_t)hs_ + c]
  float r0 = HS(y0, x0) * (1.0f - wx) + HS(y0, x1) * wx;
  float r1 = HS(y1, x0) * (1.0f - wx) + HS(y1, x1) * wx;
#undef HS
  float v = r0 * (1.0f - wy) + r1 * wy;
  out[p * (size_t)os_ + c] += q[p * (size_t)qs_ + qo_ + c] * v * cof[b * 6 + ci];
}

// ---------------------------------------------------------------------------
// Gate network: stage 1 partial max/sum per (b, chunk, c); 144 chunks of 256px
// ---------------------------------------------------------------------------
__global__ void k_pool_partial(const float* __restrict__ q, int qs_, int qo_,
                               float* __restrict__ pmax, float* __restrict__ psum) {
  int b = blockIdx.x / 144;
  int chunk = blockIdx.x % 144;
  int c = threadIdx.x;  // 64
  float mx = -3.4e38f, sm = 0.0f;
  int base = chunk * 256;
  for (int i = 0; i < 256; ++i) {
    float v = q[((size_t)b * HW_ + base + i) * (size_t)qs_ + qo_ + c];
    mx = fmaxf(mx, v);
    sm += v;
  }
  pmax[(b * 144 + chunk) * 64 + c] = mx;
  psum[(b * 144 + chunk) * 64 + c] = sm;
}

// stage 2: finish pooling, fc0/fc1, noisy top-3, masked softmax -> cof[b][6]
__global__ void k_gate_finalize(const float* __restrict__ pmax,
                                const float* __restrict__ psum,
                                const float* __restrict__ fc0w, const float* __restrict__ fc0b,
                                const float* __restrict__ fc1w, const float* __restrict__ fc1b,
                                float* __restrict__ cof) {
  int b = blockIdx.x;
  int c = threadIdx.x;  // 64
  float mx = -3.4e38f, sm = 0.0f;
  for (int ch = 0; ch < 144; ++ch) {
    mx = fmaxf(mx, pmax[(b * 144 + ch) * 64 + c]);
    sm += psum[(b * 144 + ch) * 64 + c];
  }
  __shared__ float pool[64];
  __shared__ float hh[6], nz[6];
  pool[c] = mx + sm * (1.0f / HW_);
  __syncthreads();
  if (c < 6) {
    float h = fc1b[c], n = fc0b[c];
    for (int k = 0; k < 64; ++k) {
      h += pool[k] * fc1w[c * 64 + k];
      n += pool[k] * fc0w[c * 64 + k];
    }
    hh[c] = (h > 0.0f) ? h : 0.2f * h;                 // leaky_relu 0.2
    nz[c] = (n > 20.0f) ? n : log1pf(expf(n));          // softplus
  }
  __syncthreads();
  if (c == 0) {
    float nm = 0.0f;
    for (int e = 0; e < 6; ++e) nm += nz[e];
    nm *= (1.0f / 6.0f);
    float ss = 0.0f;
    for (int e = 0; e < 6; ++e) { float d = nz[e] - nm; ss += d * d; }
    float ns = sqrtf(ss / 5.0f);                        // ddof=1
    float score[6];
    for (int e = 0; e < 6; ++e) score[e] = hh[e] + (nz[e] - nm) / ns;
    int sel[3];
    bool used[6] = {false, false, false, false, false, false};
    for (int t = 0; t < 3; ++t) {
      int bi = 0; float bv = -3.4e38f;
      for (int e = 0; e < 6; ++e)
        if (!used[e] && score[e] > bv) { bv = score[e]; bi = e; }
      used[bi] = true; sel[t] = bi;
    }
    float m = -3.4e38f;
    for (int t = 0; t < 3; ++t) m = fmaxf(m, hh[sel[t]]);
    float ex[3], den = 0.0f;
    for (int t = 0; t < 3; ++t) { ex[t] = expf(hh[sel[t]] - m); den += ex[t]; }
    for (int e = 0; e < 6; ++e) cof[b * 6 + e] = 0.0f;
    for (int t = 0; t < 3; ++t) cof[b * 6 + sel[t]] = ex[t] / den;
  }
}

// ---------------------------------------------------------------------------
// Host orchestration
// ---------------------------------------------------------------------------
extern "C" void kernel_launch(void* const* d_in, const int* in_sizes, int n_in,
                              void* d_out, int out_size, void* d_ws, size_t ws_size,
                              hipStream_t stream) {
  (void)in_sizes; (void)n_in; (void)out_size; (void)ws_size;

  // d_in flattening per jax pytree sorted-key order:
  // 0:x, experts[0..1].focal[3], experts[2..5]{c1_b,c1_w,c2_b,c2_w,e_b,e_w,focal[3]},
  // fn1_b, fn1_w, fn2_b, fn2_w, gate{fc0_b,fc0_w,fc1_b,fc1_w}, gate_b, gate_w,
  // ln1_b, ln1_w, ln2_b, ln2_w, proj1_b, proj1_w, proj2_b, proj2_w
  const float* x = (const float*)d_in[0];
  const float* spa0_f[3] = {(const float*)d_in[1], (const float*)d_in[2], (const float*)d_in[3]};
  const float* spa1_f[3] = {(const float*)d_in[4], (const float*)d_in[5], (const float*)d_in[6]};
  struct EIdx { int c1b, c1w, c2b, c2w, eb, ew, f3, f5, f7, d; };
  const EIdx eidx[4] = {
      {7, 8, 9, 10, 11, 12, 13, 14, 15, 32},
      {16, 17, 18, 19, 20, 21, 22, 23, 24, 16},
      {25, 26, 27, 28, 29, 30, 31, 32, 33, 8},
      {34, 35, 36, 37, 38, 39, 40, 41, 42, 4}};
  const float* fn1_b  = (const float*)d_in[43];
  const float* fn1_w  = (const float*)d_in[44];
  const float* fn2_b  = (const float*)d_in[45];
  const float* fn2_w  = (const float*)d_in[46];
  const float* fc0_b  = (const float*)d_in[47];
  const float* fc0_w  = (const float*)d_in[48];
  const float* fc1_b  = (const float*)d_in[49];
  const float* fc1_w  = (const float*)d_in[50];
  const float* gate_b = (const float*)d_in[51];
  const float* gate_w = (const float*)d_in[52];
  const float* ln1_b  = (const float*)d_in[53];
  const float* ln1_w  = (const float*)d_in[54];
  const float* ln2_b  = (const float*)d_in[55];
  const float* ln2_w  = (const float*)d_in[56];
  const float* p1_b   = (const float*)d_in[57];
  const float* p1_w   = (const float*)d_in[58];
  const float* p2_b   = (const float*)d_in[59];
  const float* p2_w   = (const float*)d_in[60];

  float* ws = (float*)d_ws;
  size_t o = 0;
  float* XP  = ws;      o += (size_t)P_ * 132;  // x_q|x_v|gates NHWC; later fn1 output (P*128)
  float* XN  = ws + o;  o += (size_t)P_ * 64;   // ln1 out; later ln2 out
  float* CT1 = ws + o;  o += (size_t)P_ * 64;   // ctx tmp; later Q1|Q2; later dwg
  float* CT2 = ws + o;  o += (size_t)P_ * 64;   // ctx tmp; later Q3|Q4; later a*dwg
  float* ACC = ws + o;  o += (size_t)P_ * 64;   // ctx_all; later 4 half-res buffers
  float* OUTB = ws + o; o += (size_t)P_ * 64;   // expert accumulator
  float* X1  = ws + o;  o += (size_t)P_ * 64;   // proj2 + x
  float* HG  = ws + o;  o += (size_t)P2_ * 4;   // half-res gates
  float* PM  = ws + o;  o += (size_t)B_ * 144 * 64;
  float* PS  = ws + o;  o += (size_t)B_ * 144 * 64;
  float* COF = ws + o;  o += 64;
  float* MEAN = ws + o; o += (size_t)B_ * 64;
  _Float16* WH = (_Float16*)(ws + o);

  float* Hbuf = XP;                              // fn1 output alias (P*128 <= P*132)
  float* HV = ACC;
  float* HA = ACC + (size_t)P2_ * 64;
  float* HB = HA + (size_t)P2_ * 64;
  float* HC = HB + (size_t)P2_ * 64;
  float* Q1 = CT1;
  float* Q2 = CT1 + (size_t)P_ * 32;
  float* Q3 = CT2;
  float* Q4 = CT2 + (size_t)P_ * 32;

  // f16 swizzled weight areas
  _Float16* Wp1 = WH;
  _Float16* Wp2 = Wp1 + 9 * 2 * 512;
  _Float16* Wf1 = Wp2 + 4 * 2 * 512;
  _Float16* Wf2 = Wf1 + 8 * 2 * 512;
  _Float16* cur = Wf2 + 4 * 2 * 512;
  _Float16 *Wc1[4], *Wc2[4], *We[4];
  for (int i = 0; i < 4; ++i) {
    int nt = (eidx[i].d + 15) / 16;
    Wc1[i] = cur; cur += (size_t)nt * 2 * 512;
    Wc2[i] = cur; cur += (size_t)nt * 2 * 512;
    We[i]  = cur; cur += (size_t)4 * 1 * 512;
  }

  auto prep = [&](const float* src, _Float16* dst, int Cout, int Cin, int nt, int ks) {
    k_prep_w<<<dim3(nt * ks), dim3(32), 0, stream>>>(src, dst, Cout, Cin, nt, ks);
  };
  const dim3 cgrid((P_ / 16) / 8), cblk(256);
  auto conv = [&](const float* in, int is, int io, int Cin, const _Float16* wt,
                  const float* bias, int Cout, int nt, int ks,
                  float* out, int os, int oo, int flags,
                  const float* cof, int ci, const float* resid, int rs, int ro) {
    if (ks == 2 && nt == 9)
      k_conv1x1_wmma<2, 9><<<cgrid, cblk, 0, stream>>>(in, is, io, Cin, wt, bias, Cout, out, os, oo, flags, cof, ci, resid, rs, ro);
    else if (ks == 2 && nt == 8)
      k_conv1x1_wmma<2, 8><<<cgrid, cblk, 0, stream>>>(in, is, io, Cin, wt, bias, Cout, out, os, oo, flags, cof, ci, resid, rs, ro);
    else if (ks == 2 && nt == 4)
      k_conv1x1_wmma<2, 4><<<cgrid, cblk, 0, stream>>>(in, is, io, Cin, wt, bias, Cout, out, os, oo, flags, cof, ci, resid, rs, ro);
    else if (ks == 2 && nt == 2)
      k_conv1x1_wmma<2, 2><<<cgrid, cblk, 0, stream>>>(in, is, io, Cin, wt, bias, Cout, out, os, oo, flags, cof, ci, resid, rs, ro);
    else if (ks == 2 && nt == 1)
      k_conv1x1_wmma<2, 1><<<cgrid, cblk, 0, stream>>>(in, is, io, Cin, wt, bias, Cout, out, os, oo, flags, cof, ci, resid, rs, ro);
    else  // (ks == 1 && nt == 4)
      k_conv1x1_wmma<1, 4><<<cgrid, cblk, 0, stream>>>(in, is, io, Cin, wt, bias, Cout, out, os, oo, flags, cof, ci, resid, rs, ro);
  };
  auto dwc = [&](const float* in, int is, int io, const float* w, const float* bias,
                 float* out, int os, int oo, int Cd, int Hd, int Wd, int K, int gelu) {
    size_t total = (size_t)B_ * Hd * Wd * Cd;
    k_dwconv<<<dim3((unsigned)((total + 255) / 256)), dim3(256), 0, stream>>>(
        in, is, io, w, bias, out, os, oo, Cd, Hd, Wd, K, gelu, total);
  };
  auto mulg = [&](const float* in, int is, int io, const float* gate, int gs, int gl,
                  float* out, int os, int oo, int Cd, int accum, size_t npixall) {
    size_t total = npixall * Cd;
    k_mulgate<<<dim3((unsigned)((total + 255) / 256)), dim3(256), 0, stream>>>(
        in, is, io, gate, gs, gl, out, os, oo, Cd, accum, total);
  };

  // ---- weight prep (every call; deterministic) ----
  prep(p1_w, Wp1, 132, 64, 9, 2);
  prep(p2_w, Wp2, 64, 64, 4, 2);
  prep(fn1_w, Wf1, 128, 64, 8, 2);
  prep(fn2_w, Wf2, 64, 64, 4, 2);
  for (int i = 0; i < 4; ++i) {
    int d = eidx[i].d, nt = (d + 15) / 16;
    prep((const float*)d_in[eidx[i].c1w], Wc1[i], d, 64, nt, 2);
    prep((const float*)d_in[eidx[i].c2w], Wc2[i], d, 64, nt, 2);
    prep((const float*)d_in[eidx[i].ew],  We[i], 64, d, 4, 1);
  }

  // ---- 1. LN1 (NCHW -> NHWC) ----
  k_ln_nchw2nhwc<<<dim3((P_ + 255) / 256), dim3(256), 0, stream>>>(x, ln1_w, ln1_b, XN);

  // ---- 2. proj1: 64 -> 132 (x_q | x_v | gates) ----
  conv(XN, 64, 0, 64, Wp1, p1_b, 132, 9, 2, XP, 132, 0, 0, nullptr, 0, nullptr, 0, 0);

  // ---- 3. gate network -> COF ----
  k_pool_partial<<<dim3(B_ * 144), dim3(64), 0, stream>>>(XP, 132, 0, PM, PS);
  k_gate_finalize<<<dim3(B_), dim3(64), 0, stream>>>(PM, PS, fc0_w, fc0_b, fc1_w, fc1_b, COF);

  // ---- 4. spa expert 0 (full res) ----
  dwc(XP, 132, 64, spa0_f[0], nullptr, CT1, 64, 0, 64, H_, W_, 3, 1);
  mulg(CT1, 64, 0, XP, 132, 128, ACC, 64, 0, 64, 0, P_);
  dwc(CT1, 64, 0, spa0_f[1], nullptr, CT2, 64, 0, 64, H_, W_, 5, 1);
  mulg(CT2, 64, 0, XP, 132, 129, ACC, 64, 0, 64, 1, P_);
  dwc(CT2, 64, 0, spa0_f[2], nullptr, CT1, 64, 0, 64, H_, W_, 7, 1);
  mulg(CT1, 64, 0, XP, 132, 130, ACC, 64, 0, 64, 1, P_);
  k_rowmean<<<dim3(B_ * 64), dim3(256), 0, stream>>>(CT1, 64, 0, 64, HW_, MEAN);
  {
    size_t total = (size_t)P_ * 64;
    k_combine<<<dim3((unsigned)((total + 255) / 256)), dim3(256), 0, stream>>>(
        ACC, 64, 0, MEAN, XP, 132, 131, XP, 132, 0, COF, 0,
        OUTB, 64, 0, 64, HW_, MB_MULQ | MB_COF, total);
  }

  // ---- 5. spa expert 1 (half res) ----
  {
    size_t th = (size_t)P2_ * 64;
    k_avgpool2<<<dim3((unsigned)((th + 255) / 256)), dim3(256), 0, stream>>>(
        XP, 132, 64, HV, 64, 0, 64, H2_, W2_, th);
    size_t tg = (size_t)P2_ * 4;
    k_avgpool2<<<dim3((unsigned)((tg + 255) / 256)), dim3(256), 0, stream>>>(
        XP, 132, 128, HG, 4, 0, 4, H2_, W2_, tg);
  }
  dwc(HV, 64, 0, spa1_f[0], nullptr, HA, 64, 0, 64, H2_, W2_, 3, 1);
  mulg(HA, 64, 0, HG, 4, 0, HC, 64, 0, 64, 0, P2_);
  dwc(HA, 64, 0, spa1_f[1], nullptr, HB, 64, 0, 64, H2_, W2_, 5, 1);
  mulg(HB, 64, 0, HG, 4, 1, HC, 64, 0, 64, 1, P2_);
  dwc(HB, 64, 0, spa1_f[2], nullptr, HA, 64, 0, 64, H2_, W2_, 7, 1);
  mulg(HA, 64, 0, HG, 4, 2, HC, 64, 0, 64, 1, P2_);
  k_rowmean<<<dim3(B_ * 64), dim3(256), 0, stream>>>(HA, 64, 0, 64, HW2_, MEAN);
  {
    size_t th = (size_t)P2_ * 64;
    k_combine<<<dim3((unsigned)((th + 255) / 256)), dim3(256), 0, stream>>>(
        HC, 64, 0, MEAN, HG, 4, 3, nullptr, 0, 0, nullptr, 0,
        HB, 64, 0, 64, HW2_, 0, th);
    size_t tf = (size_t)P_ * 64;
    k_upsample_acc<<<dim3((unsigned)((tf + 255) / 256)), dim3(256), 0, stream>>>(
        HB, 64, XP, 132, 0, COF, 1, OUTB, 64, tf);
  }

  // ---- 6. channel experts ----
  for (int i = 0; i < 4; ++i) {
    int d = eidx[i].d, nt = (d + 15) / 16;
    const float* c1b = (const float*)d_in[eidx[i].c1b];
    const float* c2b = (const float*)d_in[eidx[i].c2b];
    const float* ebi = (const float*)d_in[eidx[i].eb];
    const float* f3 = (const float*)d_in[eidx[i].f3];
    const float* f5 = (const float*)d_in[eidx[i].f5];
    const float* f7 = (const float*)d_in[eidx[i].f7];
    conv(XP, 132, 0, 64, Wc1[i], c1b, d, nt, 2, Q1, 32, 0, 0, nullptr, 0, nullptr, 0, 0);
    conv(XP, 132, 64, 64, Wc2[i], c2b, d, nt, 2, Q2, 32, 0, 0, nullptr, 0, nullptr, 0, 0);
    dwc(Q2, 32, 0, f3, nullptr, Q3, 32, 0, d, H_, W_, 3, 1);
    mulg(Q3, 32, 0, XP, 132, 128, Q4, 32, 0, d, 0, P_);
    dwc(Q3, 32, 0, f5, nullptr, Q2, 32, 0, d, H_, W_, 5, 1);
    mulg(Q2, 32, 0, XP, 132, 129, Q4, 32, 0, d, 1, P_);
    dwc(Q2, 32, 0, f7, nullptr, Q3, 32, 0, d, H_, W_, 7, 1);
    mulg(Q3, 32, 0, XP, 132, 130, Q4, 32, 0, d, 1, P_);
    k_rowmean<<<dim3(B_ * d), dim3(256), 0, stream>>>(Q3, 32, 0, d, HW_, MEAN);
    size_t total = (size_t)P_ * d;
    k_combine<<<dim3((unsigned)((total + 255) / 256)), dim3(256), 0, stream>>>(
        Q4, 32, 0, MEAN, XP, 132, 131, Q1, 32, 0, nullptr, 0,
        Q2, 32, 0, d, HW_, MB_MULQ, total);
    conv(Q2, 32, 0, d, We[i], ebi, 64, 4, 1, OUTB, 64, 0,
         CF_ACCUM | CF_SCALE, COF, 2 + i, nullptr, 0, 0);
  }

  // ---- 7. proj2 + x residual ----
  conv(OUTB, 64, 0, 64, Wp2, p2_b, 64, 4, 2, X1, 64, 0,
       CF_ADD_NCHW, nullptr, 0, x, 0, 0);

  // ---- 8. LN2, fn1 (GELU fused) ----
  k_ln_nhwc<<<dim3((P_ + 255) / 256), dim3(256), 0, stream>>>(X1, 64, ln2_w, ln2_b, XN);
  conv(XN, 64, 0, 64, Wf1, fn1_b, 128, 8, 2, Hbuf, 128, 0,
       CF_GELU, nullptr, 0, nullptr, 0, 0);

  // ---- 9. gated branch: dw3x3(g)+bias, a * dwg ----
  dwc(Hbuf, 128, 64, gate_w, gate_b, CT1, 64, 0, 64, H_, W_, 3, 0);
  {
    size_t total = (size_t)P_ * 64;
    k_ewmul<<<dim3((unsigned)((total + 255) / 256)), dim3(256), 0, stream>>>(
        Hbuf, 128, 0, CT1, 64, 0, CT2, 64, 0, 64, total);
  }

  // ---- 10. fn2 (GELU) + x1 residual, NCHW store to d_out ----
  conv(CT2, 64, 0, 64, Wf2, fn2_b, 64, 4, 2, (float*)d_out, 0, 0,
       CF_GELU | CF_OUT_NCHW | CF_ADD_NHWC, nullptr, 0, X1, 64, 0);
}
